// NaiveLSTM_52630529245782
// MI455X (gfx1250) — matrix operand order repaired and checked by reading, not compiled, not run on hardware
//
#include <hip/hip_runtime.h>

// ---------------- CDNA5 wave32 WMMA types ----------------
typedef __attribute__((ext_vector_type(16))) __bf16          v16bf;
typedef __attribute__((ext_vector_type(8)))  float           v8f;
typedef __attribute__((ext_vector_type(8)))  unsigned short  v8us;

#define B_   64
#define T_   256
#define D_   1024
#define H_   1024
#define G4_  4096   // 4*H

union Frag16 {
    v16bf v;
    struct { v8us lo; v8us hi; } p;
};

__device__ __forceinline__ unsigned short f2bf(float f) {
    unsigned int x = __float_as_uint(f);
    unsigned int r = x + 0x7FFFu + ((x >> 16) & 1u);   // round-to-nearest-even
    return (unsigned short)(r >> 16);
}

// A fragment: 16x32 bf16, row-major A with leading dim lda (global or LDS).
// ISA 7.12.2 (16-bit A 16x32): lanes 0-15 row M=lane, K = k0+0..7 & k0+16..23
//                              lanes 16-31 row M=lane-16, K = k0+8..15 & k0+24..31
__device__ __forceinline__ v16bf load_a_frag(const unsigned short* __restrict__ A,
                                             int lda, int m_base, int k0, int lane) {
    int half = lane >> 4;
    int row  = m_base + (lane & 15);
    const unsigned short* p = A + (size_t)row * lda + k0 + half * 8;
    Frag16 f;
    f.p.lo = *(const v8us*)(p);        // K +0..7   (or +8..15)
    f.p.hi = *(const v8us*)(p + 16);   // K +16..23 (or +24..31)
    return f.v;
}

// B fragment: 32x16 bf16 from N-major (transposed) weights Bt[N][K].
// lanes 0-15: col N=lane, K=k0+0..7 & k0+16..23 ; lanes 16-31: col N=lane-16, K=k0+8..15 & k0+24..31
__device__ __forceinline__ v16bf load_b_frag(const unsigned short* __restrict__ Bt,
                                             int ldb, int n_base, int k0, int lane) {
    int half = lane >> 4;
    int col  = n_base + (lane & 15);
    const unsigned short* p = Bt + (size_t)col * ldb + k0 + half * 8;
    Frag16 f;
    f.p.lo = *(const v8us*)(p);        // K +0..7   (or +8..15)
    f.p.hi = *(const v8us*)(p + 16);   // K +16..23 (or +24..31)
    return f.v;
}

#define WMMA_BF16(acc, a, b) \
    acc = __builtin_amdgcn_wmma_f32_16x16x32_bf16(false, (a), false, (b), (short)0, (acc), false, false)

// ---------------- pre-pass conversion kernels ----------------
__global__ void cvt_bf16_kernel(const float* __restrict__ src,
                                unsigned short* __restrict__ dst, int n) {
    int i = blockIdx.x * blockDim.x + threadIdx.x;
    if (i < n) dst[i] = f2bf(src[i]);
}

// dst[4H][K] bf16, dst[n][d] = src_{n/H}[d][n%H]  (K == H_ == 1024)
__global__ void pack_gates_T_kernel(const float* __restrict__ s0, const float* __restrict__ s1,
                                    const float* __restrict__ s2, const float* __restrict__ s3,
                                    unsigned short* __restrict__ dst) {
    int idx = blockIdx.x * blockDim.x + threadIdx.x;   // 0 .. 4H*1024
    int n   = idx >> 10;
    int d   = idx & 1023;
    int g   = n >> 10;
    int col = n & 1023;
    const float* s = (g == 0) ? s0 : (g == 1) ? s1 : (g == 2) ? s2 : s3;
    dst[idx] = f2bf(s[(size_t)d * H_ + col]);
}

// Wy_t[O][H] = W_y[H][O]
__global__ void pack_wy_T_kernel(const float* __restrict__ src, unsigned short* __restrict__ dst) {
    int idx = blockIdx.x * blockDim.x + threadIdx.x;   // 0 .. O*H
    int o = idx >> 10;
    int h = idx & 1023;
    dst[idx] = f2bf(src[(size_t)h * 1024 + o]);
}

__global__ void init_state_kernel(const float* __restrict__ h0, const float* __restrict__ c0,
                                  unsigned short* __restrict__ h, float* __restrict__ c, int n) {
    int i = blockIdx.x * blockDim.x + threadIdx.x;
    if (i < n) { h[i] = f2bf(h0[i]); c[i] = c0[i]; }
}

// ---------------- GEMM1: gx[t][b][4H] = x @ W^T + b ----------------
// Block: 256 thr = 8 waves, all share one 32-row A panel (LDS-staged, 64KB).
// Each wave: 2 M-tiles x 16 cols x 4 gates = 8 accumulators.
// B fragments ping-pong (k-loop step 64): no register copies, no WMMA->VALU
// hazard NOPs in the stream. Grid: 512 mgroups x 8 ngroups = 4096 blocks.
__global__ void __launch_bounds__(256)
gemm_gx_kernel(const unsigned short* __restrict__ xb,
               const unsigned short* __restrict__ Wt,
               const float* __restrict__ bi, const float* __restrict__ bf,
               const float* __restrict__ bj, const float* __restrict__ bo,
               float* __restrict__ gx) {
    __shared__ __attribute__((aligned(16))) unsigned short lds_a[32 * D_];
    int waveid = threadIdx.x >> 5;
    int lane   = threadIdx.x & 31;
    int m0     = (blockIdx.x >> 3) << 5;                       // 32-row group
    int n_base = (((blockIdx.x & 7) << 3) + waveid) << 4;      // 16-col tile

    // cooperative stage: 32 contiguous rows of A -> LDS
    {
        const v8us* src = (const v8us*)(xb + (size_t)m0 * D_);
        v8us*       dst = (v8us*)lds_a;
        for (int i = threadIdx.x; i < (32 * D_) / 8; i += 256) dst[i] = src[i];
    }
    __syncthreads();

    v8f z = {};
    v8f acc[8];                                   // [gate*2 + mtile]
#pragma unroll
    for (int i = 0; i < 8; ++i) acc[i] = z;

    v16bf bA[4], bB[4];
#pragma unroll
    for (int g = 0; g < 4; ++g) bA[g] = load_b_frag(Wt, D_, g * H_ + n_base, 0, lane);

    for (int k0 = 0; k0 < D_; k0 += 64) {
        // stage 1: consume bA, prefetch bB for k0+32
#pragma unroll
        for (int g = 0; g < 4; ++g) bB[g] = load_b_frag(Wt, D_, g * H_ + n_base, k0 + 32, lane);
        {
            v16bf a0 = load_a_frag(lds_a, D_, 0, k0, lane);
            v16bf a1 = load_a_frag(lds_a, D_, 16, k0, lane);
#pragma unroll
            for (int g = 0; g < 4; ++g) {
                WMMA_BF16(acc[g * 2 + 0], a0, bA[g]);
                WMMA_BF16(acc[g * 2 + 1], a1, bA[g]);
            }
        }
        // stage 2: consume bB, prefetch bA for k0+64 (wrap on last iter)
        int k2 = (k0 + 64) & (D_ - 1);
#pragma unroll
        for (int g = 0; g < 4; ++g) bA[g] = load_b_frag(Wt, D_, g * H_ + n_base, k2, lane);
        {
            v16bf a0 = load_a_frag(lds_a, D_, 0, k0 + 32, lane);
            v16bf a1 = load_a_frag(lds_a, D_, 16, k0 + 32, lane);
#pragma unroll
            for (int g = 0; g < 4; ++g) {
                WMMA_BF16(acc[g * 2 + 0], a0, bB[g]);
                WMMA_BF16(acc[g * 2 + 1], a1, bB[g]);
            }
        }
    }

    int half = lane >> 4;
    int n = n_base + (lane & 15);
    float vb0 = bi[n], vb1 = bf[n], vb2 = bj[n], vb3 = bo[n];
#pragma unroll
    for (int mt = 0; mt < 2; ++mt) {
#pragma unroll
        for (int r = 0; r < 8; ++r) {
            int row = m0 + mt * 16 + half * 8 + r;   // = b*T + t
            int bb = row >> 8;
            int tt = row & 255;
            float* out = gx + ((size_t)tt * B_ + bb) * G4_;
            out[0 * H_ + n] = acc[0 + mt][r] + vb0;
            out[1 * H_ + n] = acc[2 + mt][r] + vb1;
            out[2 * H_ + n] = acc[4 + mt][r] + vb2;
            out[3 * H_ + n] = acc[6 + mt][r] + vb3;
        }
    }
}

// ---------------- recurrent step: g = gx_t + h@U ; gates ; c,h update ----------------
// Block: 8 waves sharing one 16-row h panel (LDS, 32KB); each wave 1 n-tile x 4 gates.
// U fragments ping-pong. Grid: 4 m-tiles x 8 ngroups = 32 blocks x 256 thr.
__global__ void __launch_bounds__(256)
lstm_step_kernel(const unsigned short* __restrict__ h_prev,
                 const unsigned short* __restrict__ Ut,
                 const float* __restrict__ gx_t,
                 float* __restrict__ c,
                 unsigned short* __restrict__ h_next,
                 unsigned short* __restrict__ hs, int t) {
    __shared__ __attribute__((aligned(16))) unsigned short lds_h[16 * H_];
    int waveid = threadIdx.x >> 5;
    int lane   = threadIdx.x & 31;
    int m_base = (blockIdx.x >> 3) << 4;                       // batch tile
    int n_base = (((blockIdx.x & 7) << 3) + waveid) << 4;      // hidden tile

    {
        const v8us* src = (const v8us*)(h_prev + (size_t)m_base * H_);
        v8us*       dst = (v8us*)lds_h;
        for (int i = threadIdx.x; i < (16 * H_) / 8; i += 256) dst[i] = src[i];
    }
    __syncthreads();

    v8f z = {};
    v8f acc[4];
#pragma unroll
    for (int i = 0; i < 4; ++i) acc[i] = z;

    v16bf bA[4], bB[4];
#pragma unroll
    for (int g = 0; g < 4; ++g) bA[g] = load_b_frag(Ut, H_, g * H_ + n_base, 0, lane);

    for (int k0 = 0; k0 < H_; k0 += 64) {
#pragma unroll
        for (int g = 0; g < 4; ++g) bB[g] = load_b_frag(Ut, H_, g * H_ + n_base, k0 + 32, lane);
        {
            v16bf a = load_a_frag(lds_h, H_, 0, k0, lane);
#pragma unroll
            for (int g = 0; g < 4; ++g) WMMA_BF16(acc[g], a, bA[g]);
        }
        int k2 = (k0 + 64) & (H_ - 1);
#pragma unroll
        for (int g = 0; g < 4; ++g) bA[g] = load_b_frag(Ut, H_, g * H_ + n_base, k2, lane);
        {
            v16bf a = load_a_frag(lds_h, H_, 0, k0 + 32, lane);
#pragma unroll
            for (int g = 0; g < 4; ++g) WMMA_BF16(acc[g], a, bB[g]);
        }
    }

    int half = lane >> 4;
    int n = n_base + (lane & 15);
#pragma unroll
    for (int r = 0; r < 8; ++r) {
        int m = m_base + half * 8 + r;                 // batch index
        const float* g = gx_t + (size_t)m * G4_;
        float gi = acc[0][r] + g[0 * H_ + n];
        float gf = acc[1][r] + g[1 * H_ + n];
        float gj = acc[2][r] + g[2 * H_ + n];
        float go = acc[3][r] + g[3 * H_ + n];
        float iv = 1.0f / (1.0f + __expf(-gi));
        float fv = 1.0f / (1.0f + __expf(-gf));
        float jv = tanhf(gj);
        float ov = 1.0f / (1.0f + __expf(-go));
        size_t ci = (size_t)m * H_ + n;
        float cn = fv * c[ci] + iv * jv;
        c[ci] = cn;
        unsigned short hb = f2bf(ov * tanhf(cn));
        h_next[ci] = hb;
        hs[((size_t)m * T_ + t) * H_ + n] = hb;        // [B, T, H]
    }
}

// ---------------- GEMM3: y = hs @ W_y + b_y ----------------
// Block: 512 thr = 16 waves sharing one 32-row A panel (LDS, 64KB).
// Each wave: 2 M-tiles x 4 N-tiles (64 cols) = 8 accs, B ping-pong. Grid: 512 blocks.
__global__ void __launch_bounds__(512)
gemm_y_kernel(const unsigned short* __restrict__ hsb,
              const unsigned short* __restrict__ Wyt,
              const float* __restrict__ by,
              float* __restrict__ y) {
    __shared__ __attribute__((aligned(16))) unsigned short lds_a[32 * H_];
    int waveid = threadIdx.x >> 5;        // 0..15 -> column group
    int lane   = threadIdx.x & 31;
    int m0     = blockIdx.x << 5;         // 32-row group
    int n0     = waveid << 6;             // 64-col group

    {
        const v8us* src = (const v8us*)(hsb + (size_t)m0 * H_);
        v8us*       dst = (v8us*)lds_a;
        for (int i = threadIdx.x; i < (32 * H_) / 8; i += 512) dst[i] = src[i];
    }
    __syncthreads();

    v8f z = {};
    v8f acc[8];                           // [ntile*2 + mtile]
#pragma unroll
    for (int i = 0; i < 8; ++i) acc[i] = z;

    v16bf bA[4], bB[4];
#pragma unroll
    for (int nt = 0; nt < 4; ++nt) bA[nt] = load_b_frag(Wyt, H_, n0 + nt * 16, 0, lane);

    for (int k0 = 0; k0 < H_; k0 += 64) {
#pragma unroll
        for (int nt = 0; nt < 4; ++nt) bB[nt] = load_b_frag(Wyt, H_, n0 + nt * 16, k0 + 32, lane);
        {
            v16bf a0 = load_a_frag(lds_a, H_, 0, k0, lane);
            v16bf a1 = load_a_frag(lds_a, H_, 16, k0, lane);
#pragma unroll
            for (int nt = 0; nt < 4; ++nt) {
                WMMA_BF16(acc[nt * 2 + 0], a0, bA[nt]);
                WMMA_BF16(acc[nt * 2 + 1], a1, bA[nt]);
            }
        }
        int k2 = (k0 + 64) & (H_ - 1);
#pragma unroll
        for (int nt = 0; nt < 4; ++nt) bA[nt] = load_b_frag(Wyt, H_, n0 + nt * 16, k2, lane);
        {
            v16bf a0 = load_a_frag(lds_a, H_, 0, k0 + 32, lane);
            v16bf a1 = load_a_frag(lds_a, H_, 16, k0 + 32, lane);
#pragma unroll
            for (int nt = 0; nt < 4; ++nt) {
                WMMA_BF16(acc[nt * 2 + 0], a0, bB[nt]);
                WMMA_BF16(acc[nt * 2 + 1], a1, bB[nt]);
            }
        }
    }

    int half = lane >> 4;
    int nl = lane & 15;
#pragma unroll
    for (int mt = 0; mt < 2; ++mt) {
#pragma unroll
        for (int r = 0; r < 8; ++r) {
            int row = m0 + mt * 16 + half * 8 + r;     // = b*T + t
            float* out = y + (size_t)row * 1024;
#pragma unroll
            for (int nt = 0; nt < 4; ++nt) {
                int n = n0 + nt * 16 + nl;
                out[n] = acc[nt * 2 + mt][r] + by[n];
            }
        }
    }
}

// ---------------- host launcher ----------------
extern "C" void kernel_launch(void* const* d_in, const int* in_sizes, int n_in,
                              void* d_out, int out_size, void* d_ws, size_t ws_size,
                              hipStream_t stream) {
    (void)in_sizes; (void)n_in; (void)out_size; (void)ws_size;
    const float* x   = (const float*)d_in[0];
    const float* h0  = (const float*)d_in[1];
    const float* c0  = (const float*)d_in[2];
    const float* W_i = (const float*)d_in[3];
    const float* W_f = (const float*)d_in[4];
    const float* W_j = (const float*)d_in[5];
    const float* W_o = (const float*)d_in[6];
    const float* U_i = (const float*)d_in[7];
    const float* U_f = (const float*)d_in[8];
    const float* U_j = (const float*)d_in[9];
    const float* U_o = (const float*)d_in[10];
    const float* b_i = (const float*)d_in[11];
    const float* b_f = (const float*)d_in[12];
    const float* b_j = (const float*)d_in[13];
    const float* b_o = (const float*)d_in[14];
    const float* W_y = (const float*)d_in[15];
    const float* b_y = (const float*)d_in[16];
    float* y = (float*)d_out;

    char* ws = (char*)d_ws;
    size_t off = 0;
    auto carve = [&](size_t bytes) -> void* {
        void* p = ws + off;
        off += (bytes + 255) & ~(size_t)255;
        return p;
    };
    unsigned short* x_bf  = (unsigned short*)carve((size_t)B_ * T_ * D_ * 2);   // 32 MB
    unsigned short* W_t   = (unsigned short*)carve((size_t)G4_ * D_ * 2);       //  8 MB
    unsigned short* U_t   = (unsigned short*)carve((size_t)G4_ * H_ * 2);       //  8 MB
    unsigned short* Wy_t  = (unsigned short*)carve((size_t)1024 * H_ * 2);      //  2 MB
    float*          gx    = (float*)carve((size_t)T_ * B_ * G4_ * 4);           // 256 MB
    unsigned short* hsbuf = (unsigned short*)carve((size_t)B_ * T_ * H_ * 2);   // 32 MB
    unsigned short* hbuf0 = (unsigned short*)carve((size_t)B_ * H_ * 2);
    unsigned short* hbuf1 = (unsigned short*)carve((size_t)B_ * H_ * 2);
    float*          cbuf  = (float*)carve((size_t)B_ * H_ * 4);

    // 1) convert / pack
    {
        int n = B_ * T_ * D_;
        cvt_bf16_kernel<<<(n + 255) / 256, 256, 0, stream>>>(x, x_bf, n);
    }
    pack_gates_T_kernel<<<(G4_ * D_) / 256, 256, 0, stream>>>(W_i, W_f, W_j, W_o, W_t);
    pack_gates_T_kernel<<<(G4_ * H_) / 256, 256, 0, stream>>>(U_i, U_f, U_j, U_o, U_t);
    pack_wy_T_kernel<<<(1024 * H_) / 256, 256, 0, stream>>>(W_y, Wy_t);
    init_state_kernel<<<(B_ * H_) / 256, 256, 0, stream>>>(h0, c0, hbuf0, cbuf, B_ * H_);

    // 2) gx = x @ W + b : 512 mgroups x 8 ngroups
    gemm_gx_kernel<<<4096, 256, 0, stream>>>(x_bf, W_t, b_i, b_f, b_j, b_o, gx);

    // 3) recurrence: 256 dependent small GEMM+gate kernels (ping-pong h)
    unsigned short* hA = hbuf0;
    unsigned short* hB = hbuf1;
    for (int t = 0; t < T_; ++t) {
        lstm_step_kernel<<<32, 256, 0, stream>>>(hA, U_t, gx + (size_t)t * B_ * G4_,
                                                 cbuf, hB, hsbuf, t);
        unsigned short* tmp = hA; hA = hB; hB = tmp;
    }

    // 4) y = hs @ W_y + b_y : 512 blocks x 16 waves
    gemm_y_kernel<<<512, 512, 0, stream>>>(hsbuf, Wy_t, b_y, y);
}
